// Backbone_26121991094980
// MI455X (gfx1250) — matrix operand order repaired and checked by reading, not compiled, not run on hardware
//
#include <hip/hip_runtime.h>
#include <hip/hip_bf16.h>
#include <stdint.h>

typedef _Float16 half_t;
typedef __attribute__((ext_vector_type(16))) _Float16 v16h;
typedef __attribute__((ext_vector_type(8)))  _Float16 v8h;
typedef __attribute__((ext_vector_type(8)))  float    v8f;
typedef __attribute__((ext_vector_type(4)))  unsigned int v4u;
typedef __attribute__((ext_vector_type(8)))  int      v8i;
typedef __attribute__((ext_vector_type(4)))  int      v4i;

#define EPSBN 1e-5f
#define INV_SQRT_DM 0.04419417382415922f   // 1/sqrt(512)

// ---------------------------------------------------------------------------
// WMMA helpers (wave32, 16x16x32 f16 -> f32)
// ---------------------------------------------------------------------------
__device__ inline v8f wmma32(v16h a, v16h b, v8f c) {
  return __builtin_amdgcn_wmma_f32_16x16x32_f16(false, a, false, b, (short)0, c,
                                                false, false);
}

// A fragment: 16x32 f16 tile, row-major, stride ld halves. Rows must be
// 16B-aligned (ld even, base 16B aligned). Lane l<16 -> row l, K {0..7,16..23};
// lane>=16 -> row l-16, K {8..15,24..31}.  Two contiguous 8-half runs.
__device__ inline v16h frag_a_full(const half_t* A, int ld, int lane) {
  int r  = lane & 15;
  int hi = lane >> 4;
  const half_t* p = A + (size_t)r * ld + (hi ? 8 : 0);
  v8h lo = *(const v8h*)p;
  v8h hh = *(const v8h*)(p + 16);
  return __builtin_shufflevector(lo, hh, 0, 1, 2, 3, 4, 5, 6, 7,
                                 8, 9, 10, 11, 12, 13, 14, 15);
}

// B fragment from Bt stored N-major: Bt[n*ld + k] == B[k][n].
// Lane l<16 -> col l, K {0..7,16..23}; lane>=16 -> col l-16, K {8..15,24..31}.
__device__ inline v16h frag_b_full(const half_t* Bt, int ld, int lane) {
  int c  = lane & 15;
  int hi = lane >> 4;
  const half_t* p = Bt + (size_t)c * ld + (hi ? 8 : 0);
  v8h lo = *(const v8h*)p;
  v8h hh = *(const v8h*)(p + 16);
  return __builtin_shufflevector(lo, hh, 0, 1, 2, 3, 4, 5, 6, 7,
                                 8, 9, 10, 11, 12, 13, 14, 15);
}

// ---------------------------------------------------------------------------
// Generic WMMA GEMM: C[M,N] = act(bn(scale*A[M,Kd] @ Bt[N,Kd]^T + bias)) + resid
// A row stride == Kd. 64x32 C tile per block, 8 waves, each wave one 16x16 tile.
// ---------------------------------------------------------------------------
template <typename TA>
__global__ void __launch_bounds__(256) gemm_bt_kernel(
    const TA* __restrict__ A, long long strideA,
    const half_t* __restrict__ Bt, int ldb, long long strideB,
    int M, int N, int Kd,
    float* __restrict__ Cf, half_t* __restrict__ Ch,
    int ldc, long long strideC, int outT, int ldcT,
    float scale,
    const float* __restrict__ bias,
    const float* __restrict__ bng, const float* __restrict__ bnb,
    const float* __restrict__ bnm, const float* __restrict__ bnv,
    int relu,
    const float* __restrict__ resid, int ldres, long long strideR) {
  __shared__ alignas(16) half_t As[64 * 40];
  __shared__ alignas(16) half_t Bs[32 * 40];
  int tid  = threadIdx.x;
  int wave = tid >> 5;
  int lane = tid & 31;
  int wr = wave >> 1;   // 0..3
  int wc = wave & 1;    // 0..1
  int bm = blockIdx.y * 64;
  int bn = blockIdx.x * 32;
  const TA*     Ab = A  + (size_t)blockIdx.z * strideA;
  const half_t* Bb = Bt + (size_t)blockIdx.z * strideB;

  bool fullM = (bm + 64 <= M);
  bool fullN = (bn + 32 <= N);

  v8f acc = {0.f, 0.f, 0.f, 0.f, 0.f, 0.f, 0.f, 0.f};

  for (int k0 = 0; k0 < Kd; k0 += 32) {
    bool fullK = (k0 + 32 <= Kd);
    if (fullM && fullK) {
      for (int i = tid; i < 64 * 32; i += 256) {
        int r = i >> 5, kk = i & 31;
        As[r * 40 + kk] = (half_t)(float)Ab[(size_t)(bm + r) * Kd + k0 + kk];
      }
    } else {
      for (int i = tid; i < 64 * 32; i += 256) {
        int r = i >> 5, kk = i & 31;
        int gm = bm + r, gk = k0 + kk;
        float v = (gm < M && gk < Kd) ? (float)Ab[(size_t)gm * Kd + gk] : 0.f;
        As[r * 40 + kk] = (half_t)v;
      }
    }
    if (fullN && fullK) {
      for (int i = tid; i < 32 * 32; i += 256) {
        int r = i >> 5, kk = i & 31;
        Bs[r * 40 + kk] = (half_t)(float)Bb[(size_t)(bn + r) * ldb + k0 + kk];
      }
    } else {
      for (int i = tid; i < 32 * 32; i += 256) {
        int r = i >> 5, kk = i & 31;
        int gn = bn + r, gk = k0 + kk;
        float v = (gn < N && gk < Kd) ? (float)Bb[(size_t)gn * ldb + gk] : 0.f;
        Bs[r * 40 + kk] = (half_t)v;
      }
    }
    if (k0 + 32 < Kd)  // speculative prefetch of next A chunk -> global_prefetch_b8
      __builtin_prefetch(Ab + (size_t)(bm + (tid >> 2)) * Kd + k0 + 32, 0, 1);
    __syncthreads();
    v16h a = frag_a_full(As + (wr * 16) * 40, 40, lane);
    v16h b = frag_b_full(Bs + (wc * 16) * 40, 40, lane);
    acc = wmma32(a, b, acc);
    __syncthreads();
  }

  int c  = lane & 15;
  int hi = lane >> 4;
#pragma unroll
  for (int e = 0; e < 8; ++e) {
    int m = bm + wr * 16 + e + (hi ? 8 : 0);
    int n = bn + wc * 16 + c;
    if (m >= M || n >= N) continue;
    float x = acc[e] * scale;
    if (bias) x += bias[n];
    if (bng)  x = (x - bnm[n]) * bng[n] * rsqrtf(bnv[n] + EPSBN) + bnb[n];
    if (relu) x = fmaxf(x, 0.f);
    if (resid) x += resid[(size_t)blockIdx.z * strideR + (size_t)m * ldres + n];
    size_t co = (size_t)blockIdx.z * strideC;
    if (Cf) Cf[co + (size_t)m * ldc + n] = x;
    if (Ch) {
      if (outT) Ch[co + (size_t)n * ldcT + m] = (half_t)x;
      else      Ch[co + (size_t)m * ldc  + n] = (half_t)x;
    }
  }
}

// ---------------------------------------------------------------------------
// Fused grouped transformer: one block per 16-point group (4 waves).
// feat(16,Dp) -> fc1(512) -> q,k -> softmax(q k^T/sqrt512) -> v -> P v -> fc2(Dp)+pre
// Weights pre-padded: W1p is (512, Kp) with Kp=ceil32(Dp); W2p is (ceil16(Dp), 512).
// ---------------------------------------------------------------------------
__global__ void __launch_bounds__(128) group_attn_kernel(
    const float* __restrict__ feat, float* __restrict__ out, int Dp,
    const half_t* __restrict__ W1p, const float* __restrict__ b1,
    const half_t* __restrict__ Wqh, const half_t* __restrict__ Wkh,
    const half_t* __restrict__ Wvh,
    const half_t* __restrict__ W2p, const float* __restrict__ b2) {
  // one padded shared arena; all sub-buffers 16B aligned
  __shared__ alignas(16) half_t SH[29696];
  __shared__ alignas(16) float  scoresF[256];
  half_t* featH = SH;              // 16 x Kp (Kp <= 288)
  half_t* x1H   = SH + 4608;       // 16 x 512 ; later res = P @ v
  half_t* qH    = SH + 12800;      // 16 x 512
  half_t* kH    = SH + 20992;      // 16 x 512
  half_t* pH    = SH + 29184;      // 16 x 32 (zero-padded K)
  half_t* vT32  = SH + 12800;      // 512 x 32 K-padded v^T (overlays q,k)

  int g    = blockIdx.x;
  int tid  = threadIdx.x;
  int wave = tid >> 5;
  int lane = tid & 31;
  const float* fg = feat + (size_t)g * 16 * Dp;
  int Kp = (Dp + 31) & ~31;

  for (int i = tid; i < 16 * Kp; i += 128) {
    int r = i / Kp, kk = i - r * Kp;
    featH[i] = (kk < Dp) ? (half_t)fg[(size_t)r * Dp + kk] : (half_t)0.f;
  }
  __syncthreads();

  int c  = lane & 15;
  int hi = lane >> 4;
  int ks1 = Kp >> 5;

  // x1 = feat @ W1^T + b1   (32 tiles of 16 cols)
  for (int t = wave; t < 32; t += 4) {
    v8f acc = {0.f, 0.f, 0.f, 0.f, 0.f, 0.f, 0.f, 0.f};
    for (int s = 0; s < ks1; ++s) {
      v16h a = frag_a_full(featH + s * 32, Kp, lane);
      v16h b = frag_b_full(W1p + (size_t)(t * 16) * Kp + s * 32, Kp, lane);
      acc = wmma32(a, b, acc);
    }
#pragma unroll
    for (int e = 0; e < 8; ++e) {
      int m = e + (hi ? 8 : 0);
      int n = t * 16 + c;
      x1H[m * 512 + n] = (half_t)(acc[e] + b1[n]);
    }
  }
  __syncthreads();

  // q, k = x1 @ Wq^T / Wk^T
  for (int t = wave; t < 64; t += 4) {
    const half_t* W = (t < 32) ? Wqh : Wkh;
    half_t* dst     = (t < 32) ? qH  : kH;
    int tt = t & 31;
    v8f acc = {0.f, 0.f, 0.f, 0.f, 0.f, 0.f, 0.f, 0.f};
    for (int s = 0; s < 16; ++s) {
      v16h a = frag_a_full(x1H + s * 32, 512, lane);
      v16h b = frag_b_full(W + (size_t)(tt * 16) * 512 + s * 32, 512, lane);
      acc = wmma32(a, b, acc);
    }
#pragma unroll
    for (int e = 0; e < 8; ++e)
      dst[(e + (hi ? 8 : 0)) * 512 + tt * 16 + c] = (half_t)acc[e];
  }
  __syncthreads();

  // scores = q @ k^T / sqrt(512)
  if (wave == 0) {
    v8f acc = {0.f, 0.f, 0.f, 0.f, 0.f, 0.f, 0.f, 0.f};
    for (int s = 0; s < 16; ++s) {
      v16h a = frag_a_full(qH + s * 32, 512, lane);
      v16h b = frag_b_full(kH + s * 32, 512, lane);
      acc = wmma32(a, b, acc);
    }
#pragma unroll
    for (int e = 0; e < 8; ++e)
      scoresF[(e + (hi ? 8 : 0)) * 16 + c] = acc[e] * INV_SQRT_DM;
  }
  __syncthreads();

  // row softmax (16 rows); pH zero-padded to K=32
  if (tid < 16) {
    float mx = -3.0e38f;
    for (int j = 0; j < 16; ++j) mx = fmaxf(mx, scoresF[tid * 16 + j]);
    float ex[16], sum = 0.f;
    for (int j = 0; j < 16; ++j) { ex[j] = __expf(scoresF[tid * 16 + j] - mx); sum += ex[j]; }
    float inv = 1.f / sum;
    for (int j = 0; j < 16; ++j) pH[tid * 32 + j] = (half_t)(ex[j] * inv);
    for (int j = 16; j < 32; ++j) pH[tid * 32 + j] = (half_t)0.f;
  }
  __syncthreads();

  // v = x1 @ Wv^T, stored transposed K-padded: vT32[n*32 + k], rows 16..31 zero
  for (int t = wave; t < 32; t += 4) {
    v8f acc = {0.f, 0.f, 0.f, 0.f, 0.f, 0.f, 0.f, 0.f};
    for (int s = 0; s < 16; ++s) {
      v16h a = frag_a_full(x1H + s * 32, 512, lane);
      v16h b = frag_b_full(Wvh + (size_t)(t * 16) * 512 + s * 32, 512, lane);
      acc = wmma32(a, b, acc);
    }
#pragma unroll
    for (int e = 0; e < 8; ++e) {
      int m = e + (hi ? 8 : 0);        // k index 0..15
      int n = t * 16 + c;              // 0..511
      vT32[n * 32 + m]      = (half_t)acc[e];
      vT32[n * 32 + 16 + m] = (half_t)0.f;
    }
  }
  __syncthreads();

  // res = P @ v   (into x1H)
  for (int t = wave; t < 32; t += 4) {
    v8f acc = {0.f, 0.f, 0.f, 0.f, 0.f, 0.f, 0.f, 0.f};
    v16h a = frag_a_full(pH, 32, lane);
    v16h b = frag_b_full(vT32 + (size_t)(t * 16) * 32, 32, lane);
    acc = wmma32(a, b, acc);
#pragma unroll
    for (int e = 0; e < 8; ++e)
      x1H[(e + (hi ? 8 : 0)) * 512 + t * 16 + c] = (half_t)acc[e];
  }
  __syncthreads();

  // out = res @ W2^T + b2 + feat     (W2p rows padded to ceil16(Dp))
  int ntiles = (Dp + 15) >> 4;
  for (int t = wave; t < ntiles; t += 4) {
    v8f acc = {0.f, 0.f, 0.f, 0.f, 0.f, 0.f, 0.f, 0.f};
    for (int s = 0; s < 16; ++s) {
      v16h a = frag_a_full(x1H + s * 32, 512, lane);
      v16h b = frag_b_full(W2p + (size_t)(t * 16) * 512 + s * 32, 512, lane);
      acc = wmma32(a, b, acc);
    }
#pragma unroll
    for (int e = 0; e < 8; ++e) {
      int m = e + (hi ? 8 : 0);
      int n = t * 16 + c;
      if (n < Dp)
        out[(size_t)g * 16 * Dp + (size_t)m * Dp + n] =
            acc[e] + b2[n] + fg[(size_t)m * Dp + n];
    }
  }
}

// ---------------------------------------------------------------------------
// TDM probe: exercise the Tensor Data Mover (async tensor load -> LDS) with a
// deterministic 64x16 f32 tile copy. Output goes to workspace scratch.
// ---------------------------------------------------------------------------
__global__ void tdm_probe_kernel(const float* __restrict__ src, float* __restrict__ dst) {
  __shared__ alignas(16) float buf[64 * 16];
#if __has_builtin(__builtin_amdgcn_tensor_load_to_lds)
  unsigned long long ga = (unsigned long long)(uintptr_t)src;
  v4u g0;
  g0[0] = 1u;                                   // count=1, user descriptor
  g0[1] = 0u;                                   // lds_addr = 0 (start of LDS alloc)
  g0[2] = (unsigned)(ga & 0xFFFFFFFFu);         // global_addr[31:0]
  g0[3] = (unsigned)((ga >> 32) & 0x1FFFFFFu) | (2u << 30);  // addr[56:32], type=2
  v8i g1;
  g1[0] = (int)(2u << 16);                      // data_size = 4B
  g1[1] = (int)(64u << 16);                     // tensor_dim0 = 64 (low16 @ bits63:48)
  g1[2] = (int)(16u << 16);                     // tensor_dim1 = 16 (low16 @ bits95:80)
  g1[3] = (int)(64u << 16);                     // tile_dim0 = 64 (bits127:112)
  g1[4] = 16;                                   // tile_dim1 = 16 (bits143:128)
  g1[5] = 64;                                   // tensor_dim0_stride = 64
  g1[6] = 0;
  g1[7] = 0;
  v4i gz; gz[0] = 0; gz[1] = 0; gz[2] = 0; gz[3] = 0;
#if __clang_major__ >= 23
  v8i gz8; gz8[0]=0; gz8[1]=0; gz8[2]=0; gz8[3]=0; gz8[4]=0; gz8[5]=0; gz8[6]=0; gz8[7]=0;
  __builtin_amdgcn_tensor_load_to_lds(g0, g1, gz, gz, gz8, 0);
#else
  __builtin_amdgcn_tensor_load_to_lds(g0, g1, gz, gz, 0);
#endif
  __builtin_amdgcn_s_wait_tensorcnt(0);
  __syncthreads();
  if (threadIdx.x == 0) dst[0] = buf[0] + buf[63];
#else
  if (threadIdx.x == 0) dst[0] = src[0];
  (void)buf;
#endif
}

// ---------------------------------------------------------------------------
// Supporting kernels
// ---------------------------------------------------------------------------
__global__ void cvt_kernel(const float* __restrict__ s, half_t* __restrict__ d, size_t n) {
  size_t i = (size_t)blockIdx.x * 256 + threadIdx.x;
  if (i < n) d[i] = (half_t)s[i];
}

// padded f32 -> f16 matrix convert: d is (rowsPad, colsPad), zero outside (rows, cols)
__global__ void cvt_pad_kernel(const float* __restrict__ s, half_t* __restrict__ d,
                               int rows, int cols, int rowsPad, int colsPad) {
  int t = blockIdx.x * 256 + threadIdx.x;
  int total = rowsPad * colsPad;
  if (t >= total) return;
  int r = t / colsPad, k = t - r * colsPad;
  d[t] = (r < rows && k < cols) ? (half_t)s[(size_t)r * cols + k] : (half_t)0.f;
}

__global__ void extract_xyz_kernel(const float* __restrict__ x, float* __restrict__ xyz, int npts) {
  int t = blockIdx.x * blockDim.x + threadIdx.x;
  if (t >= npts) return;
  xyz[t * 3 + 0] = x[t * 6 + 0];
  xyz[t * 3 + 1] = x[t * 6 + 1];
  xyz[t * 3 + 2] = x[t * 6 + 2];
}

// top-16 nearest-neighbor selection (stable on ties, like argsort)
__global__ void knn_kernel(const float* __restrict__ q, const float* __restrict__ r,
                           int Bb, int M, int N, int* __restrict__ out) {
  int t = blockIdx.x * blockDim.x + threadIdx.x;
  if (t >= Bb * M) return;
  int b = t / M;
  const float* qp = q + (size_t)t * 3;
  float cx = qp[0], cy = qp[1], cz = qp[2];
  const float* rb = r + (size_t)b * N * 3;
  float bd[16]; int bi[16];
#pragma unroll
  for (int j = 0; j < 16; ++j) { bd[j] = 3.0e38f; bi[j] = 0; }
  for (int n = 0; n < N; ++n) {
    float dx = rb[n * 3] - cx, dy = rb[n * 3 + 1] - cy, dz = rb[n * 3 + 2] - cz;
    float d = dx * dx + dy * dy + dz * dz;
    if (d < bd[15]) {
      int pos = 15;
      for (int j = 14; j >= 0; --j) { if (d < bd[j]) pos = j; else break; }
      for (int j = 15; j > pos; --j) { bd[j] = bd[j - 1]; bi[j] = bi[j - 1]; }
      bd[pos] = d; bi[pos] = n;
    }
  }
  int* o = out + (size_t)t * 16;
#pragma unroll
  for (int j = 0; j < 16; ++j) o[j] = bi[j];
}

__global__ void build_feat0_kernel(const float* __restrict__ x, const float* __restrict__ xyz,
                                   const int* __restrict__ idx, float* __restrict__ out,
                                   int N, int count) {
  int t = blockIdx.x * blockDim.x + threadIdx.x;
  if (t >= count) return;
  int pm = t / 16;           // b*N + n
  int b  = pm / N;
  int id = idx[t];
  const float* nb = x + ((size_t)b * N + id) * 6;
  const float* ce = xyz + (size_t)pm * 3;
  float* o = out + (size_t)t * 9;
  o[0] = ce[0] - nb[0]; o[1] = ce[1] - nb[1]; o[2] = ce[2] - nb[2];
  for (int cc = 0; cc < 6; ++cc) o[3 + cc] = nb[cc];
}

__global__ void build_feat_sa_kernel(const float* __restrict__ xyz, const float* __restrict__ nxyz,
                                     const float* __restrict__ pts, const int* __restrict__ idx,
                                     float* __restrict__ out, int Ncur, int np, int Dp, int count) {
  int t = blockIdx.x * blockDim.x + threadIdx.x;
  if (t >= count) return;
  int pm = t / 16;           // b*np + m
  int b  = pm / np;
  int id = idx[t];
  const float* nb = xyz + ((size_t)b * Ncur + id) * 3;
  const float* ce = nxyz + (size_t)pm * 3;
  int oc = 3 + Dp;
  float* o = out + (size_t)t * oc;
  o[0] = nb[0] - ce[0]; o[1] = nb[1] - ce[1]; o[2] = nb[2] - ce[2];
  const float* pp = pts + ((size_t)b * Ncur + id) * Dp;
  for (int cc = 0; cc < Dp; ++cc) o[3 + cc] = pp[cc];
}

__global__ void gather3_kernel(const float* __restrict__ xyz, const int* __restrict__ idx,
                               float* __restrict__ out, int Ncur, int np, int count) {
  int t = blockIdx.x * blockDim.x + threadIdx.x;
  if (t >= count) return;
  int b  = t / np;
  int id = idx[t];
  const float* s = xyz + ((size_t)b * Ncur + id) * 3;
  out[t * 3 + 0] = s[0]; out[t * 3 + 1] = s[1]; out[t * 3 + 2] = s[2];
}

__global__ void maxpool16_kernel(const float* __restrict__ in, float* __restrict__ out,
                                 int rows, int C) {
  int t = blockIdx.x * blockDim.x + threadIdx.x;
  if (t >= rows * C) return;
  int row = t / C, cc = t - row * C;
  const float* p = in + (size_t)row * 16 * C + cc;
  float m = p[0];
  for (int j = 1; j < 16; ++j) m = fmaxf(m, p[(size_t)j * C]);
  out[t] = m;
}

__global__ void __launch_bounds__(256) softmax_rows_kernel(const float* __restrict__ S,
                                                           half_t* __restrict__ P, int W) {
  __shared__ float red[256];
  size_t row = blockIdx.x;
  const float* s = S + row * (size_t)W;
  half_t* p = P + row * (size_t)W;
  int tid = threadIdx.x;
  float mx = -3.0e38f;
  for (int j = tid; j < W; j += 256) mx = fmaxf(mx, s[j]);
  red[tid] = mx; __syncthreads();
  for (int st = 128; st > 0; st >>= 1) {
    if (tid < st) red[tid] = fmaxf(red[tid], red[tid + st]);
    __syncthreads();
  }
  mx = red[0]; __syncthreads();
  float sum = 0.f;
  for (int j = tid; j < W; j += 256) sum += __expf(s[j] - mx);
  red[tid] = sum; __syncthreads();
  for (int st = 128; st > 0; st >>= 1) {
    if (tid < st) red[tid] += red[tid + st];
    __syncthreads();
  }
  float inv = 1.f / red[0];
  for (int j = tid; j < W; j += 256) p[j] = (half_t)(__expf(s[j] - mx) * inv);
}

// farthest point sampling; one block per batch; records far, updates dmin, argmax
__global__ void __launch_bounds__(256) fps_kernel(const float* __restrict__ xyz,
                                                  int N, int npoint, int* __restrict__ out) {
  __shared__ float dmin[2048];
  __shared__ float rv[256];
  __shared__ int   ri[256];
  int b = blockIdx.x;
  int tid = threadIdx.x;
  const float* xb = xyz + (size_t)b * N * 3;
  for (int n = tid; n < N; n += 256) dmin[n] = 1.0e10f;
  __syncthreads();
  int far = 0;
  for (int it = 0; it < npoint; ++it) {
    if (tid == 0) out[b * npoint + it] = far;
    float cx = xb[far * 3], cy = xb[far * 3 + 1], cz = xb[far * 3 + 2];
    float bv = -1.f; int bI = 0;
    for (int n = tid; n < N; n += 256) {
      float dx = xb[n * 3] - cx, dy = xb[n * 3 + 1] - cy, dz = xb[n * 3 + 2] - cz;
      float d = dx * dx + dy * dy + dz * dz;
      float dm = fminf(dmin[n], d);
      dmin[n] = dm;
      if (dm > bv) { bv = dm; bI = n; }
    }
    rv[tid] = bv; ri[tid] = bI;
    __syncthreads();
    for (int st = 128; st > 0; st >>= 1) {
      if (tid < st) {
        if (rv[tid + st] > rv[tid] ||
            (rv[tid + st] == rv[tid] && ri[tid + st] < ri[tid])) {
          rv[tid] = rv[tid + st]; ri[tid] = ri[tid + st];
        }
      }
      __syncthreads();
    }
    far = ri[0];
    __syncthreads();
  }
}

// ---------------------------------------------------------------------------
// Host orchestration
// ---------------------------------------------------------------------------
struct BnP  { const float *b, *g, *m, *v; };
struct LinP { const float *b, *w; };
struct TbP  { LinP fc1, fc2; const float *wk, *wq, *wv; };
struct BlkP { BnP bn1, bn2; LinP conv1, conv2; TbP sa_t, t; };

static void launch_gemm(hipStream_t stream, const void* A, bool aHalf, long long sA,
                        const half_t* Bt, int ldb, long long sB,
                        int M, int N, int Kd,
                        float* Cf, half_t* Ch, int ldc, long long sC, int outT, int ldcT,
                        float scale, const float* bias, const BnP* bn, int relu,
                        const float* resid, int ldres, long long sR, int batch) {
  dim3 grid((N + 31) / 32, (M + 63) / 64, batch);
  const float* g = bn ? bn->g : nullptr;
  const float* b = bn ? bn->b : nullptr;
  const float* m = bn ? bn->m : nullptr;
  const float* v = bn ? bn->v : nullptr;
  if (aHalf)
    gemm_bt_kernel<half_t><<<grid, dim3(256), 0, stream>>>(
        (const half_t*)A, sA, Bt, ldb, sB, M, N, Kd, Cf, Ch, ldc, sC, outT, ldcT,
        scale, bias, g, b, m, v, relu, resid, ldres, sR);
  else
    gemm_bt_kernel<float><<<grid, dim3(256), 0, stream>>>(
        (const float*)A, sA, Bt, ldb, sB, M, N, Kd, Cf, Ch, ldc, sC, outT, ldcT,
        scale, bias, g, b, m, v, relu, resid, ldres, sR);
}

extern "C" void kernel_launch(void* const* d_in, const int* in_sizes, int n_in,
                              void* d_out, int out_size, void* d_ws, size_t ws_size,
                              hipStream_t stream) {
  const int Bc = 2, N = 2048;

  // ---- parameter pytree walk (JAX flattens dicts by sorted key) ----
  int pi = 1;  // d_in[0] == x
  auto nextp = [&]() { return (const float*)d_in[pi++]; };
  auto readBn  = [&]() { BnP r;  r.b = nextp(); r.g = nextp(); r.m = nextp(); r.v = nextp(); return r; };
  auto readLin = [&]() { LinP r; r.b = nextp(); r.w = nextp(); return r; };
  auto readTb  = [&]() { TbP r;  r.fc1 = readLin(); r.fc2 = readLin();
                         r.wk = nextp(); r.wq = nextp(); r.wv = nextp(); return r; };
  BlkP blocks[4];
  for (int i = 0; i < 4; ++i) {  // keys: bn1, bn2, conv1, conv2, sa_t, t
    blocks[i].bn1 = readBn(); blocks[i].bn2 = readBn();
    blocks[i].conv1 = readLin(); blocks[i].conv2 = readLin();
    blocks[i].sa_t = readTb(); blocks[i].t = readTb();
  }
  BnP  fc_delta_bn  = readBn();
  LinP fc_delta_lin = readLin();
  BnP  linear1_bn   = readBn();
  LinP linear1_lin  = readLin();
  TbP  t0 = readTb();
  TbP  t1 = readTb();

  // ---- workspace bump allocator ----
  size_t off = 0;
  auto alloc = [&](size_t bytes) -> void* {
    void* p = (char*)d_ws + off;
    off += (bytes + 255) & ~(size_t)255;
    return p;
  };
  auto toH = [&](const float* w, size_t n) -> half_t* {
    half_t* d = (half_t*)alloc(n * sizeof(half_t));
    cvt_kernel<<<dim3((unsigned)((n + 255) / 256)), dim3(256), 0, stream>>>(w, d, n);
    return d;
  };
  auto toHpad = [&](const float* w, int rows, int cols, int rowsPad, int colsPad) -> half_t* {
    half_t* d = (half_t*)alloc((size_t)rowsPad * colsPad * sizeof(half_t));
    int total = rowsPad * colsPad;
    cvt_pad_kernel<<<dim3((total + 255) / 256), dim3(256), 0, stream>>>(
        w, d, rows, cols, rowsPad, colsPad);
    return d;
  };

  // TDM probe (exercises tensor_load_to_lds + s_wait_tensorcnt)
  float* tdmDst = (float*)alloc(256);
  tdm_probe_kernel<<<dim3(1), dim3(32), 0, stream>>>((const float*)d_in[0], tdmDst);

  // ---- global transformer (attention over all Mp points) ----
  auto xformer = [&](const float* fin, float* fout, int Bb, int Mp, int D, const TbP& tb) {
    half_t* W1h = toH(tb.fc1.w, (size_t)512 * D);
    half_t* Wqh = toH(tb.wq, 512 * 512);
    half_t* Wkh = toH(tb.wk, 512 * 512);
    half_t* Wvh = toH(tb.wv, 512 * 512);
    half_t* W2h = toH(tb.fc2.w, (size_t)D * 512);
    int Mtot = Bb * Mp;
    half_t* x1 = (half_t*)alloc((size_t)Mtot * 512 * 2);
    launch_gemm(stream, fin, false, 0, W1h, D, 0, Mtot, 512, D,
                nullptr, x1, 512, 0, 0, 0, 1.f, tb.fc1.b, nullptr, 0, nullptr, 0, 0, 1);
    half_t* qh = (half_t*)alloc((size_t)Mtot * 512 * 2);
    launch_gemm(stream, x1, true, 0, Wqh, 512, 0, Mtot, 512, 512,
                nullptr, qh, 512, 0, 0, 0, 1.f, nullptr, nullptr, 0, nullptr, 0, 0, 1);
    half_t* kh = (half_t*)alloc((size_t)Mtot * 512 * 2);
    launch_gemm(stream, x1, true, 0, Wkh, 512, 0, Mtot, 512, 512,
                nullptr, kh, 512, 0, 0, 0, 1.f, nullptr, nullptr, 0, nullptr, 0, 0, 1);
    half_t* vT = (half_t*)alloc((size_t)Mtot * 512 * 2);  // per-batch (512, Mp)
    launch_gemm(stream, x1, true, (long long)Mp * 512, Wvh, 512, 0, Mp, 512, 512,
                nullptr, vT, 512, (long long)512 * Mp, 1, Mp,
                1.f, nullptr, nullptr, 0, nullptr, 0, 0, Bb);
    float* sc = (float*)alloc((size_t)Bb * Mp * Mp * 4);
    launch_gemm(stream, qh, true, (long long)Mp * 512, kh, 512, (long long)Mp * 512,
                Mp, Mp, 512, sc, nullptr, Mp, (long long)Mp * Mp, 0, 0,
                INV_SQRT_DM, nullptr, nullptr, 0, nullptr, 0, 0, Bb);
    half_t* ph = (half_t*)alloc((size_t)Bb * Mp * Mp * 2);
    softmax_rows_kernel<<<dim3((unsigned)(Bb * Mp)), dim3(256), 0, stream>>>(sc, ph, Mp);
    half_t* resh = (half_t*)alloc((size_t)Mtot * 512 * 2);
    launch_gemm(stream, ph, true, (long long)Mp * Mp, vT, Mp, (long long)512 * Mp,
                Mp, 512, Mp, nullptr, resh, 512, (long long)Mp * 512, 0, 0,
                1.f, nullptr, nullptr, 0, nullptr, 0, 0, Bb);
    launch_gemm(stream, resh, true, 0, W2h, 512, 0, Mtot, D, 512,
                fout, nullptr, D, 0, 0, 0, 1.f, tb.fc2.b, nullptr, 0, fin, D, 0, 1);
  };

  // ---- grouped transformer (attention within each 16-point group) ----
  auto gxformer = [&](const float* fin, float* fout, int ngroups, int Dp, const TbP& tb) {
    int Kp  = (Dp + 31) & ~31;
    int Np2 = (Dp + 15) & ~15;
    half_t* W1p = toHpad(tb.fc1.w, 512, Dp, 512, Kp);
    half_t* Wqh = toH(tb.wq, 512 * 512);
    half_t* Wkh = toH(tb.wk, 512 * 512);
    half_t* Wvh = toH(tb.wv, 512 * 512);
    half_t* W2p = toHpad(tb.fc2.w, Dp, 512, Np2, 512);
    group_attn_kernel<<<dim3((unsigned)ngroups), dim3(128), 0, stream>>>(
        fin, fout, Dp, W1p, tb.fc1.b, Wqh, Wkh, Wvh, W2p, tb.fc2.b);
  };

  const float* x = (const float*)d_in[0];

  // xyz = x[..., :3]
  float* xyz = (float*)alloc((size_t)Bc * N * 3 * 4);
  extract_xyz_kernel<<<dim3((Bc * N + 127) / 128), dim3(128), 0, stream>>>(x, xyz, Bc * N);

  // knn over all points, build t = [center-nbr, knn_x]
  int* knn0 = (int*)alloc((size_t)Bc * N * 16 * 4);
  knn_kernel<<<dim3((Bc * N + 127) / 128), dim3(128), 0, stream>>>(xyz, xyz, Bc, N, N, knn0);
  float* feat0 = (float*)alloc((size_t)Bc * N * 16 * 9 * 4);
  build_feat0_kernel<<<dim3((Bc * N * 16 + 127) / 128), dim3(128), 0, stream>>>(
      x, xyz, knn0, feat0, N, Bc * N * 16);

  // h = transformer(t, t0)
  float* h0 = (float*)alloc((size_t)Bc * N * 16 * 9 * 4);
  gxformer(feat0, h0, Bc * N, 9, t0);

  // fc_delta + bn + relu, then max over group
  half_t* Wfd = toH(fc_delta_lin.w, (size_t)32 * 9);
  float* fd = (float*)alloc((size_t)Bc * N * 16 * 32 * 4);
  launch_gemm(stream, h0, false, 0, Wfd, 9, 0, Bc * N * 16, 32, 9,
              fd, nullptr, 32, 0, 0, 0, 1.f, fc_delta_lin.b, &fc_delta_bn, 1,
              nullptr, 0, 0, 1);
  float* hp = (float*)alloc((size_t)Bc * N * 32 * 4);
  maxpool16_kernel<<<dim3((Bc * N * 32 + 127) / 128), dim3(128), 0, stream>>>(fd, hp, Bc * N, 32);

  // linear1 + bn + relu
  half_t* Wl1 = toH(linear1_lin.w, (size_t)32 * 32);
  float* h1 = (float*)alloc((size_t)Bc * N * 32 * 4);
  launch_gemm(stream, hp, false, 0, Wl1, 32, 0, Bc * N, 32, 32,
              h1, nullptr, 32, 0, 0, 0, 1.f, linear1_lin.b, &linear1_bn, 1,
              nullptr, 0, 0, 1);

  // points = transformer(h, t1)  (global attention over 2048 points)
  float* pts = (float*)alloc((size_t)Bc * N * 32 * 4);
  xformer(h1, pts, Bc, N, 32, t1);

  // ---- SA blocks ----
  int Ncur = N, Dpts = 32;
  float* xyzc = xyz;
  for (int i = 0; i < 4; ++i) {
    int np = N / (4 << (2 * i));   // 512, 128, 32, 8
    int ch = 64 << i;              // 64, 128, 256, 512
    int in_ch = Dpts + 3;

    int* fpsIdx = (int*)alloc((size_t)Bc * np * 4);
    fps_kernel<<<dim3(Bc), dim3(256), 0, stream>>>(xyzc, Ncur, np, fpsIdx);

    float* nxyz = (float*)alloc((size_t)Bc * np * 3 * 4);
    gather3_kernel<<<dim3((Bc * np + 127) / 128), dim3(128), 0, stream>>>(
        xyzc, fpsIdx, nxyz, Ncur, np, Bc * np);

    int* idx = (int*)alloc((size_t)Bc * np * 16 * 4);
    knn_kernel<<<dim3((Bc * np + 127) / 128), dim3(128), 0, stream>>>(
        nxyz, xyzc, Bc, np, Ncur, idx);

    float* fg = (float*)alloc((size_t)Bc * np * 16 * in_ch * 4);
    build_feat_sa_kernel<<<dim3((Bc * np * 16 + 127) / 128), dim3(128), 0, stream>>>(
        xyzc, nxyz, pts, idx, fg, Ncur, np, Dpts, Bc * np * 16);

    float* fg2 = (float*)alloc((size_t)Bc * np * 16 * in_ch * 4);
    gxformer(fg, fg2, Bc * np, in_ch, blocks[i].sa_t);

    half_t* Wc1 = toH(blocks[i].conv1.w, (size_t)ch * in_ch);
    float* c1 = (float*)alloc((size_t)Bc * np * 16 * ch * 4);
    launch_gemm(stream, fg2, false, 0, Wc1, in_ch, 0, Bc * np * 16, ch, in_ch,
                c1, nullptr, ch, 0, 0, 0, 1.f, blocks[i].conv1.b, &blocks[i].bn1, 1,
                nullptr, 0, 0, 1);
    half_t* Wc2 = toH(blocks[i].conv2.w, (size_t)ch * ch);
    float* c2 = (float*)alloc((size_t)Bc * np * 16 * ch * 4);
    launch_gemm(stream, c1, false, 0, Wc2, ch, 0, Bc * np * 16, ch, ch,
                c2, nullptr, ch, 0, 0, 0, 1.f, blocks[i].conv2.b, &blocks[i].bn2, 1,
                nullptr, 0, 0, 1);

    float* pooled = (float*)alloc((size_t)Bc * np * ch * 4);
    maxpool16_kernel<<<dim3((Bc * np * ch + 127) / 128), dim3(128), 0, stream>>>(
        c2, pooled, Bc * np, ch);

    float* newPts = (i == 3) ? (float*)d_out : (float*)alloc((size_t)Bc * np * ch * 4);
    xformer(pooled, newPts, Bc, np, ch, blocks[i].t);

    pts = newPts; xyzc = nxyz; Ncur = np; Dpts = ch;
  }
  (void)in_sizes; (void)n_in; (void)out_size; (void)ws_size;
}
